// GATLayer_12730283065987
// MI455X (gfx1250) — matrix operand order repaired and checked by reading, not compiled, not run on hardware
//
#include <hip/hip_runtime.h>
#include <hip/hip_fp16.h>

typedef __attribute__((ext_vector_type(16))) _Float16 v16h;
typedef __attribute__((ext_vector_type(8)))  float    v8f;

#define N_NODES 20000
#define N_EDGES 640000
#define D       128
#define EPS_LN  1e-6f

__device__ __forceinline__ float wave_sum(float v) {
#pragma unroll
    for (int m = 16; m >= 1; m >>= 1) v += __shfl_xor(v, m, 32);
    return v;
}

// ---------------------------------------------------------------- init
// Zeros accumulators and pre-packs fc_w into WMMA B-fragment order:
//   Bpack[kblock(4)][ntile(8)][lane(32)][elem(16)]  (f16)
// For lane l, elem i of a 32-K block: k = kblock*32 + (l>>4)*16 + i,
//                                     n = ntile*16 + (l&15)          (contiguous k order, same as A)
// This makes each lane's B fragment a contiguous, 32B-aligned chunk.
__global__ void k_init(float* __restrict__ feat, float* __restrict__ esum,
                       unsigned* __restrict__ emax, _Float16* __restrict__ fcwp,
                       const float* __restrict__ fc_w) {
    int i = blockIdx.x * blockDim.x + threadIdx.x;
    if (i < N_NODES * D) feat[i] = 0.0f;
    if (i < N_NODES) { esum[i] = 0.0f; emax[i] = 0u; }   // e >= 0 -> bits(0.0) works as -inf
    if (i < 4 * 8 * 32 * 16) {
        int e    = i & 15;
        int lane = (i >> 4) & 31;
        int nt   = (i >> 9) & 7;
        int kb   = i >> 12;
        int k = kb * 32 + (lane >> 4) * 16 + e;
        int n = nt * 16 + (lane & 15);
        fcwp[i] = (_Float16)fc_w[k * D + n];
    }
}

// ------------------------------------------------- node LN + logits (eh, et)
__global__ void k_node(const float* __restrict__ h,
                       const float* __restrict__ hn_a, const float* __restrict__ hn_b,
                       const float* __restrict__ tn_a, const float* __restrict__ tn_b,
                       const float* __restrict__ head_w, const float* __restrict__ tail_w,
                       float* __restrict__ eh, float* __restrict__ et) {
    int wave = (blockIdx.x * blockDim.x + threadIdx.x) >> 5;
    int lane = threadIdx.x & 31;
    if (wave >= N_NODES) return;
    const float* x = h + (size_t)wave * D;
    float v[4]; float s = 0.0f;
#pragma unroll
    for (int j = 0; j < 4; ++j) { v[j] = x[lane + 32 * j]; s += v[j]; }
    float mean = wave_sum(s) * (1.0f / D);
    float sq = 0.0f;
#pragma unroll
    for (int j = 0; j < 4; ++j) { float d = v[j] - mean; sq += d * d; }
    float var = wave_sum(sq) * (1.0f / (D - 1));      // Bessel-corrected, per reference
    float inv = 1.0f / (sqrtf(var) + EPS_LN);
    float dh = 0.0f, dt = 0.0f;
#pragma unroll
    for (int j = 0; j < 4; ++j) {
        int c = lane + 32 * j;
        float z = (v[j] - mean) * inv;
        dh += (hn_a[c] * z + hn_b[c]) * head_w[c];
        dt += (tn_a[c] * z + tn_b[c]) * tail_w[c];
    }
    dh = wave_sum(dh); dt = wave_sum(dt);
    if (lane == 0) { eh[wave] = tanhf(dh); et[wave] = tanhf(dt); }
}

// ----------------------------- edge LN + logit e = relu(eh[src]+et[dst]+er)
// HBM-bound pass: streams r (328 MB, > 192 MB L2). One wave32 per edge.
__global__ void k_edge(const float* __restrict__ r,
                       const int* __restrict__ src, const int* __restrict__ dst,
                       const float* __restrict__ rn_a, const float* __restrict__ rn_b,
                       const float* __restrict__ rel_w,
                       const float* __restrict__ eh, const float* __restrict__ et,
                       float* __restrict__ e_raw, unsigned* __restrict__ emax) {
    int wave = (blockIdx.x * blockDim.x + threadIdx.x) >> 5;
    int lane = threadIdx.x & 31;
    if (wave >= N_EDGES) return;
    const float* x = r + (size_t)wave * D;
    float v[4]; float s = 0.0f;
#pragma unroll
    for (int j = 0; j < 4; ++j) { v[j] = x[lane + 32 * j]; s += v[j]; }
    float mean = wave_sum(s) * (1.0f / D);
    float sq = 0.0f;
#pragma unroll
    for (int j = 0; j < 4; ++j) { float d = v[j] - mean; sq += d * d; }
    float var = wave_sum(sq) * (1.0f / (D - 1));
    float inv = 1.0f / (sqrtf(var) + EPS_LN);
    float dr = 0.0f;
#pragma unroll
    for (int j = 0; j < 4; ++j) {
        int c = lane + 32 * j;
        float z = (v[j] - mean) * inv;
        dr += (rn_a[c] * z + rn_b[c]) * rel_w[c];
    }
    dr = wave_sum(dr);
    if (lane == 0) {
        float er = tanhf(dr);
        int d = dst[wave];
        float e = fmaxf(eh[src[wave]] + et[d] + er, 0.0f);
        e_raw[wave] = e;
        atomicMax(&emax[d], __float_as_uint(e));   // monotone for non-negative floats
    }
}

// ----------------------------------------- softmax: exp(e - max) + segment sum
__global__ void k_exp(float* __restrict__ ex, const unsigned* __restrict__ emax,
                      const int* __restrict__ dst, float* __restrict__ esum) {
    int i = blockIdx.x * blockDim.x + threadIdx.x;
    if (i >= N_EDGES) return;
    int d = dst[i];
    float v = __expf(ex[i] - __uint_as_float(emax[d]));
    ex[i] = v;
    atomicAdd(&esum[d], v);
}

// ----------------------------- aggregation: feat[dst] += a * h[src]  (L2 atomics)
// h (10.2 MB) and feat (10.2 MB) are L2-resident; atomics never leave GL2.
__global__ void k_scatter(const float* __restrict__ h,
                          const int* __restrict__ src, const int* __restrict__ dst,
                          const float* __restrict__ ex, const float* __restrict__ esum,
                          float* __restrict__ feat) {
    int wave = (blockIdx.x * blockDim.x + threadIdx.x) >> 5;
    int lane = threadIdx.x & 31;
    if (wave >= N_EDGES) return;
    int s = src[wave], d = dst[wave];
    float a = ex[wave] / esum[d];
    const float* hs = h + (size_t)s * D;
    float* fd = feat + (size_t)d * D;
#pragma unroll
    for (int j = 0; j < 4; ++j) {
        int c = lane + 32 * j;
        atomicAdd(&fd[c], a * hs[c]);
    }
}

// ------------------------- out = feat @ fc_w + fc_b via v_wmma_f32_16x16x32_f16
// One 16x16 tile per wave, K-loop 4 x 32. All waves fully active (EXEC all-1s).
// B fragments come pre-packed contiguous (2x global_load_b128 per K-step).
__global__ void __launch_bounds__(128)
k_gemm_wmma(const float* __restrict__ feat, const _Float16* __restrict__ fcwp,
            const float* __restrict__ fc_b, float* __restrict__ out) {
    int wave = threadIdx.x >> 5;
    int lane = threadIdx.x & 31;
    int tile = blockIdx.x * 4 + wave;          // 10000 tiles = 1250 (M) x 8 (N)
    int mt = tile >> 3;
    int nt = tile & 7;
    int half = lane >> 4;                      // K-half selector per ISA layout
    int l15  = lane & 15;
    int m_a  = mt * 16 + l15;                  // A: row per lane
    int n_b  = nt * 16 + l15;                  // B/C/D: col per lane

    // lane's packed B fragments: [kblock][nt][lane][16]
    const v16h* bfrag = (const v16h*)fcwp + (size_t)nt * 32 + lane;

    v8f acc = {};
#pragma unroll
    for (int kb = 0; kb < 4; ++kb) {
        v16h A;
        // A 16x32 f16: elem i -> K kbase + half*16 + i
        const float* arow = feat + (size_t)m_a * D + kb * 32 + half * 16;
#pragma unroll
        for (int i = 0; i < 8; ++i) {
            A[i]     = (_Float16)arow[i];
            A[i + 8] = (_Float16)arow[i + 8];
        }
        v16h B = bfrag[(size_t)kb * 8 * 32];   // contiguous 32B fragment

        acc = __builtin_amdgcn_wmma_f32_16x16x32_f16(
            /*neg_a=*/false, A, /*neg_b=*/false, B,
            /*c_mod=*/(short)0, acc, /*reuse_a=*/false, /*reuse_b=*/false);
    }
    float bias = fc_b[n_b];
#pragma unroll
    for (int j = 0; j < 8; ++j) {              // C/D: VGPR j -> row j + 8*half
        int m = mt * 16 + half * 8 + j;
        out[(size_t)m * D + n_b] = acc[j] + bias;
    }
}

// --------------------------------------------- row-wise L2 normalize, in place
__global__ void k_norm(float* __restrict__ out) {
    int wave = (blockIdx.x * blockDim.x + threadIdx.x) >> 5;
    int lane = threadIdx.x & 31;
    if (wave >= N_NODES) return;
    float* row = out + (size_t)wave * D;
    float v[4]; float s = 0.0f;
#pragma unroll
    for (int j = 0; j < 4; ++j) { v[j] = row[lane + 32 * j]; s += v[j] * v[j]; }
    s = wave_sum(s);
    float inv = 1.0f / fmaxf(sqrtf(s), 1e-12f);
#pragma unroll
    for (int j = 0; j < 4; ++j) row[lane + 32 * j] = v[j] * inv;
}

extern "C" void kernel_launch(void* const* d_in, const int* in_sizes, int n_in,
                              void* d_out, int out_size, void* d_ws, size_t ws_size,
                              hipStream_t stream) {
    const float* h      = (const float*)d_in[0];
    const float* r      = (const float*)d_in[1];
    const int*   src    = (const int*)  d_in[2];
    const int*   dst    = (const int*)  d_in[3];
    const float* hn_a   = (const float*)d_in[4];
    const float* hn_b   = (const float*)d_in[5];
    const float* tn_a   = (const float*)d_in[6];
    const float* tn_b   = (const float*)d_in[7];
    const float* rn_a   = (const float*)d_in[8];
    const float* rn_b   = (const float*)d_in[9];
    const float* head_w = (const float*)d_in[10];
    const float* tail_w = (const float*)d_in[11];
    const float* rel_w  = (const float*)d_in[12];
    const float* fc_w   = (const float*)d_in[13];
    const float* fc_b   = (const float*)d_in[14];
    float* out = (float*)d_out;

    // workspace layout (floats): eh[N] et[N] emax[N] esum[N] ex[E] feat[N*D] fcwp[D*D f16]
    float*    w    = (float*)d_ws;
    float*    eh   = w;
    float*    et   = eh + N_NODES;
    unsigned* emax = (unsigned*)(et + N_NODES);
    float*    esum = (float*)(emax + N_NODES);
    float*    ex   = esum + N_NODES;
    float*    feat = ex + N_EDGES;
    _Float16* fcwp = (_Float16*)(feat + (size_t)N_NODES * D);

    // 1) zero accumulators, pack fc_w -> f16 B-fragment order
    k_init<<<(N_NODES * D) / 256, 256, 0, stream>>>(feat, esum, emax, fcwp, fc_w);
    // 2) node logits
    k_node<<<N_NODES / 8, 256, 0, stream>>>(h, hn_a, hn_b, tn_a, tn_b, head_w, tail_w, eh, et);
    // 3) edge logits + segment max (HBM-bound: streams r)
    k_edge<<<N_EDGES / 8, 256, 0, stream>>>(r, src, dst, rn_a, rn_b, rel_w, eh, et, ex, emax);
    // 4) exp + segment sum
    k_exp<<<N_EDGES / 256, 256, 0, stream>>>(ex, emax, dst, esum);
    // 5) weighted aggregation (L2-resident atomics)
    k_scatter<<<N_EDGES / 8, 256, 0, stream>>>(h, src, dst, ex, esum, feat);
    // 6) WMMA GEMM + bias
    k_gemm_wmma<<<(N_NODES / 16) * 8 / 4, 128, 0, stream>>>(feat, fcwp, fc_b, out);
    // 7) L2 row normalize
    k_norm<<<N_NODES / 8, 256, 0, stream>>>(out);
}
